// GeometricAttentionBias_49211735277872
// MI455X (gfx1250) — matrix operand order repaired and checked by reading, not compiled, not run on hardware
//
#include <hip/hip_runtime.h>
#include <hip/hip_bf16.h>

typedef __attribute__((ext_vector_type(16))) _Float16 v16h;
typedef __attribute__((ext_vector_type(8)))  _Float16 v8h;
typedef __attribute__((ext_vector_type(2)))  _Float16 v2h;
typedef __attribute__((ext_vector_type(8)))  float    v8f;
typedef __attribute__((ext_vector_type(8)))  int      v8i;
typedef __attribute__((ext_vector_type(4)))  int      v4i;

#define NB 4
#define NG 64
#define NP 4096
#define NH 8
#define H1 128
#define H2 64
#define W2T_STRIDE 144                 // halves per row (128 + 16 pad -> clean 2-way LDS access)
#define PTS_PER_BLOCK 512
#define TILES_PER_WAVE (PTS_PER_BLOCK / (8 * 16))
#define LN_EPS 1e-5f

static __device__ __forceinline__ v8f wmma16(v16h a, v16h b, v8f c) {
  // D = A(16x32 f16) * B(32x16 f16) + C(16x16 f32)
  return __builtin_amdgcn_wmma_f32_16x16x32_f16(false, a, false, b, (short)0, c,
                                                false, false);
}

static __device__ __forceinline__ float relu1(float v) {
  // v_med3_f32 v, 0, +inf : single-instruction clamp-to-[0,inf)
  return __builtin_amdgcn_fmed3f(v, 0.0f, __builtin_inff());
}

static __device__ __forceinline__ v8f zero8() {
  v8f z;
#pragma unroll
  for (int i = 0; i < 8; ++i) z[i] = 0.0f;
  return z;
}

static __device__ __forceinline__ int pack2(float a, float b) {
  v2h p;
  p[0] = (_Float16)a;
  p[1] = (_Float16)b;          // pairs -> v_cvt_pk_f16_f32
  return __builtin_bit_cast(int, p);
}

// Build a B-fragment (K x N=points, contiguous K order used here: lane(l2,hi) half i holds
// K = kt*32 + 16*hi + i) from C-layout activations:
//   c[mt][r] = value(point = l2, feature = 16*mt + 8*his + r) held at lane (l2, his).
// Half of the needed features live in the partner half-wave -> one xor-16 exchange
// of 4 packed-f16 dwords, then a dword-granular merge.
static __device__ __forceinline__ v16h make_bfrag(const v8f* c, int kt, int hi) {
  v8f own  = c[2 * kt + hi];       // this lane's chunk of feature block 2kt+hi
  v8f send = c[2 * kt + 1 - hi];   // what the partner half-wave needs from us
  int ow[4], ru[4];
#pragma unroll
  for (int j = 0; j < 4; ++j) {
    ow[j] = pack2(own[2 * j], own[2 * j + 1]);
    int su = pack2(send[2 * j], send[2 * j + 1]);
    ru[j] = __shfl_xor(su, 16, 32);
  }
  v8i fr;
#pragma unroll
  for (int j = 0; j < 4; ++j) {
    fr[j]     = hi ? ru[j] : ow[j];   // halves i<8  <- his=0 source
    fr[4 + j] = hi ? ow[j] : ru[j];   // halves i>=8 <- his=1 source
  }
  return __builtin_bit_cast(v16h, fr);
}

__global__ __launch_bounds__(256)
void geo_bias_kernel(const float* __restrict__ scene,   // (B,NP,3)
                     const float* __restrict__ poses,   // (B,NG,7)
                     const float* __restrict__ W1, const float* __restrict__ b1,
                     const float* __restrict__ g1, const float* __restrict__ be1,
                     const float* __restrict__ W2, const float* __restrict__ b2,
                     const float* __restrict__ g2, const float* __restrict__ be2,
                     const float* __restrict__ W3, const float* __restrict__ b3,
                     const float* __restrict__ bscale,
                     float* __restrict__ out)            // (B,NH,NG,NP)
{
  __shared__ __align__(16) _Float16 w2t[H2 * W2T_STRIDE]; // gamma1-folded W2^T, f16
  __shared__ __align__(16) _Float16 w1t[H1 * 16];         // A1 rows, hi-half pre-zeroed
  __shared__ float b2p[H2];                               // b2 + beta1 @ W2
  __shared__ float b3p[NH];                               // b3 + beta2 @ W3

  const int tid  = threadIdx.x;
  const int lane = tid & 31;
  const int wv   = tid >> 5;
  const int l2   = lane & 15;   // point / A-row index
  const int hi   = lane >> 4;   // half-wave select
  const int g    = blockIdx.y;
  const int b    = blockIdx.z;

  // ---- block setup: fold LN affine into downstream weights, stage in LDS ----
  for (int idx = tid; idx < H1 * H2; idx += 256) {
    int n2 = idx & (H2 - 1);
    int k  = idx >> 6;
    w2t[n2 * W2T_STRIDE + k] = (_Float16)(g1[k] * W2[idx]);  // W2[k][n2]
  }
  // w1t row n (16 halves): [W1[0..3][n], b1[n], 0,0,0 | 0 x8]  (hi=1 K-range is all-zero)
  for (int idx = tid; idx < H1 * 16; idx += 256) {
    int n = idx >> 4, c = idx & 15;
    float v = 0.0f;
    if (c < 4)       v = W1[c * H1 + n];
    else if (c == 4) v = b1[n];                 // bias rides as K-row 4
    w1t[idx] = (_Float16)v;
  }
  if (tid < H2) {
    float s = b2[tid];
    for (int k = 0; k < H1; ++k) s += be1[k] * W2[k * H2 + tid];
    b2p[tid] = s;
  }
  if (tid < NH) {
    float s = b3[tid];
    for (int k = 0; k < H2; ++k) s += be2[k] * W3[k * NH + tid];
    b3p[tid] = s;
  }
  __syncthreads();

  // ---- per-block uniform pose -> R (applied transposed), t ----
  const float* gp = poses + ((size_t)b * NG + g) * 7;
  float tx = gp[0], ty = gp[1], tz = gp[2];
  float qx = gp[3], qy = gp[4], qz = gp[5], qw = gp[6];
  float qn = sqrtf(qx * qx + qy * qy + qz * qz + qw * qw) + 1e-8f;
  qx /= qn; qy /= qn; qz /= qn; qw /= qn;
  float r00 = 1.f - 2.f * (qy * qy + qz * qz), r01 = 2.f * (qx * qy - qw * qz), r02 = 2.f * (qx * qz + qw * qy);
  float r10 = 2.f * (qx * qy + qw * qz), r11 = 1.f - 2.f * (qx * qx + qz * qz), r12 = 2.f * (qy * qz - qw * qx);
  float r20 = 2.f * (qx * qz - qw * qy), r21 = 2.f * (qy * qz + qw * qx), r22 = 1.f - 2.f * (qx * qx + qy * qy);

  // ---- hoisted layer-3 A fragments: A3 = gamma2-folded W3^T (heads x 64K, rows>=8 zero)
  v16h a3[2];
#pragma unroll
  for (int kt = 0; kt < 2; ++kt)
#pragma unroll
    for (int i = 0; i < 16; ++i) {
      int K = kt * 32 + 16 * hi + i;
      float v = (l2 < NH) ? (g2[K] * W3[K * NH + l2]) : 0.0f;
      a3[kt][i] = (_Float16)v;
    }
  float b3v[8], sclv[8];
#pragma unroll
  for (int r = 0; r < 8; ++r) { b3v[r] = b3p[r]; sclv[r] = bscale[r]; }

  // ---- main loop: 16 points per tile per wave ----
#pragma unroll 1
  for (int t = 0; t < TILES_PER_WAVE; ++t) {
    asm volatile("" ::: "memory");  // keep per-tile LDS loads inside the loop
    const int pbase = blockIdx.x * PTS_PER_BLOCK + (wv * TILES_PER_WAVE + t) * 16;
    const int P = pbase + l2;

    // features for this lane's point (both half-waves compute the same point)
    const float* sp = scene + ((size_t)b * NP + P) * 3;
    float dx = sp[0] - tx, dy = sp[1] - ty, dz = sp[2] - tz;
    float lx = r00 * dx + r10 * dy + r20 * dz;
    float ly = r01 * dx + r11 * dy + r21 * dz;
    float lz = r02 * dx + r12 * dy + r22 * dz;
    float dist = sqrtf(lx * lx + ly * ly + lz * lz);

    // B1 = [feats^T ; ones-row] padded to K=32 (hi half-wave covers K=16..31 -> zero)
    v16h bf;
#pragma unroll
    for (int i = 0; i < 16; ++i) bf[i] = (_Float16)0.0f;
    bf[0] = (_Float16)(hi ? 0.0f : lx);
    bf[1] = (_Float16)(hi ? 0.0f : ly);
    bf[2] = (_Float16)(hi ? 0.0f : lz);
    bf[3] = (_Float16)(hi ? 0.0f : dist);
    bf[4] = (_Float16)(hi ? 0.0f : 1.0f);   // bias row

    // ---- layer 1: h1 = W1^T @ feats^T (+b1 via bias row), 8 M-tiles ----
    v8f c1[8];
#pragma unroll
    for (int mt = 0; mt < 8; ++mt) {
      const v4i row = *(const v4i*)(w1t + (mt * 16 + l2) * 16 + 8 * hi);
      v8i ai;
#pragma unroll
      for (int i = 0; i < 4; ++i) { ai[i] = row[i]; ai[4 + i] = 0; } // K>=16 -> 0
      c1[mt] = wmma16(__builtin_bit_cast(v16h, ai), bf, zero8());
    }

    // relu + LN stats (128 feats: 64 in-lane + 64 in partner half-wave)
    float s = 0.f, q = 0.f;
#pragma unroll
    for (int mt = 0; mt < 8; ++mt)
#pragma unroll
      for (int r = 0; r < 8; ++r) {
        float v = relu1(c1[mt][r]);
        c1[mt][r] = v; s += v; q += v * v;
      }
    s += __shfl_xor(s, 16, 32);
    q += __shfl_xor(q, 16, 32);
    float mu   = s * (1.0f / H1);
    float rstd = rsqrtf(q * (1.0f / H1) - mu * mu + LN_EPS);
#pragma unroll
    for (int mt = 0; mt < 8; ++mt)
#pragma unroll
      for (int r = 0; r < 8; ++r) c1[mt][r] = (c1[mt][r] - mu) * rstd;

    // ---- layer 2: h2 = (g1.W2)^T @ h1^T, 4 M-tiles x 4 K-steps ----
    v8f c2[4];
#pragma unroll
    for (int m = 0; m < 4; ++m) c2[m] = zero8();
#pragma unroll
    for (int kt = 0; kt < 4; ++kt) {
      const v16h bb = make_bfrag(c1, kt, hi);
#pragma unroll
      for (int m = 0; m < 4; ++m) {
        const _Float16* ap = w2t + (m * 16 + l2) * W2T_STRIDE + kt * 32 + 16 * hi;
        const v4i alo = *(const v4i*)ap;          // K = kt*32 + 16hi + 0..7
        const v4i ahi = *(const v4i*)(ap + 8);    // K = kt*32 + 16hi + 8..15
        v8i ai;
#pragma unroll
        for (int i = 0; i < 4; ++i) { ai[i] = alo[i]; ai[4 + i] = ahi[i]; }
        c2[m] = wmma16(__builtin_bit_cast(v16h, ai), bb, c2[m]);
      }
    }

    // +bias, relu, LN over 64 feats
    s = 0.f; q = 0.f;
#pragma unroll
    for (int m = 0; m < 4; ++m)
#pragma unroll
      for (int r = 0; r < 8; ++r) {
        float v = relu1(c2[m][r] + b2p[m * 16 + 8 * hi + r]);
        c2[m][r] = v; s += v; q += v * v;
      }
    s += __shfl_xor(s, 16, 32);
    q += __shfl_xor(q, 16, 32);
    mu   = s * (1.0f / H2);
    rstd = rsqrtf(q * (1.0f / H2) - mu * mu + LN_EPS);
#pragma unroll
    for (int m = 0; m < 4; ++m)
#pragma unroll
      for (int r = 0; r < 8; ++r) c2[m][r] = (c2[m][r] - mu) * rstd;

    // ---- layer 3: bias = (g2.W3)^T @ h2^T, heads in C rows 0..7 (hi=0 lanes) ----
    v8f c3 = zero8();
#pragma unroll
    for (int kt = 0; kt < 2; ++kt) {
      const v16h bb = make_bfrag(c2, kt, hi);
      c3 = wmma16(a3[kt], bb, c3);
    }

    if (hi == 0) {
#pragma unroll
      for (int r = 0; r < 8; ++r) {
        float v = (c3[r] + b3v[r]) * sclv[r];
        out[(((size_t)b * NH + r) * NG + g) * NP + P] = v;  // coalesced across lanes
      }
    }
  }
}

extern "C" void kernel_launch(void* const* d_in, const int* in_sizes, int n_in,
                              void* d_out, int out_size, void* d_ws, size_t ws_size,
                              hipStream_t stream) {
  // d_in[0] = grasp_tokens (unused by the reference math)
  const float* scene = (const float*)d_in[1];
  const float* poses = (const float*)d_in[2];
  const float* W1    = (const float*)d_in[3];
  const float* b1    = (const float*)d_in[4];
  const float* g1    = (const float*)d_in[5];
  const float* be1   = (const float*)d_in[6];
  const float* W2    = (const float*)d_in[7];
  const float* b2    = (const float*)d_in[8];
  const float* g2    = (const float*)d_in[9];
  const float* be2   = (const float*)d_in[10];
  const float* W3    = (const float*)d_in[11];
  const float* b3    = (const float*)d_in[12];
  const float* bs    = (const float*)d_in[13];
  float* out = (float*)d_out;

  dim3 grid(NP / PTS_PER_BLOCK, NG, NB);   // (8, 64, 4)
  geo_bias_kernel<<<grid, 256, 0, stream>>>(scene, poses, W1, b1, g1, be1,
                                            W2, b2, g2, be2, W3, b3, bs, out);
  (void)in_sizes; (void)n_in; (void)out_size; (void)d_ws; (void)ws_size;
}
